// LSTM_34359738368754
// MI455X (gfx1250) — matrix operand order repaired and checked by reading, not compiled, not run on hardware
//
#include <hip/hip_runtime.h>
#include <hip/hip_bf16.h>
#include <math.h>

// ---------------------------------------------------------------------------
// Problem constants (reference: I=108, H=1024, V=108, T=180, B=512)
// ---------------------------------------------------------------------------
#define SZ_I   108
#define SZ_IP  128   // I padded to a multiple of 32 (WMMA K step)
#define SZ_H   1024
#define SZ_G   4096  // 4*H gate width
#define SZ_V   108
#define SZ_VP  128   // V padded for full WMMA N tiles
#define SZ_T   180
#define SZ_B   512

typedef __attribute__((ext_vector_type(16))) __bf16 v16bf;
typedef __attribute__((ext_vector_type(8)))  float  v8f;

// ---------------------------------------------------------------------------
// WMMA helper: D = A(16x32 bf16) x B(32x16 bf16) + C(16x16 f32)
// ---------------------------------------------------------------------------
__device__ __forceinline__ v8f wmma_bf16(v16bf a, v16bf b, v8f c) {
  return __builtin_amdgcn_wmma_f32_16x16x32_bf16(
      /*neg_a=*/false, a, /*neg_b=*/false, b,
      /*c_mod=*/(short)0, c, /*reuse_a=*/false, /*reuse_b=*/false);
}

// Accumulate a (MF*16) x (NF*16) output tile over K.
// A: [M,K] row-major activations (rows row0..row0+MF*16-1)
// W: [N,K] row-major weights   (rows = output cols col0..col0+NF*16-1)
// ISA 16-bit A layout: lane<16 holds row (l), K {0..7,16..23}; lane>=16 K {8..15,24..31}
// ISA 16-bit B layout: lane<16 holds col (l), K {0..7,16..23}; lane>=16 K {8..15,24..31}
template <int MF, int NF>
__device__ __forceinline__ void gemm_acc(const __bf16* __restrict__ A,
                                         const __bf16* __restrict__ W,
                                         int K, int row0, int col0, int lane,
                                         v8f (&c)[MF][NF]) {
  const int l  = lane & 15;
  const int hi = lane >> 4;
  const __bf16* ap = A + (size_t)(row0 + l) * K + (size_t)hi * 8;
  const __bf16* bp = W + (size_t)(col0 + l) * K + (size_t)hi * 8;
  for (int k0 = 0; k0 < K; k0 += 32) {
    union { v16bf v; uint4 u[2]; } a[MF], b[NF];
#pragma unroll
    for (int i = 0; i < MF; ++i) {
      const __bf16* p = ap + (size_t)i * 16 * K + k0;
      a[i].u[0] = *(const uint4*)(p);
      a[i].u[1] = *(const uint4*)(p + 16);
    }
#pragma unroll
    for (int j = 0; j < NF; ++j) {
      const __bf16* p = bp + (size_t)j * 16 * K + k0;
      b[j].u[0] = *(const uint4*)(p);
      b[j].u[1] = *(const uint4*)(p + 16);
    }
#pragma unroll
    for (int i = 0; i < MF; ++i)
#pragma unroll
      for (int j = 0; j < NF; ++j)
        c[i][j] = wmma_bf16(a[i].v, b[j].v, c[i][j]);
  }
}

// ---------------------------------------------------------------------------
// GEMM: C[M,N] = A1[M,K1] * W1[N,K1]^T (+ A2[M,K2] * W2[N,K2]^T) + bias[N]
// Block = 256 threads = 8 waves (2 M-waves x 4 N-waves).
// Each wave computes 32x64 (2x4 fragments) -> block tile 64 x 256.
// Grid: (N/256, M/64). OUT_BF16 selects output type.
// ---------------------------------------------------------------------------
template <bool OUT_BF16>
__global__ __launch_bounds__(256) void gemm2_kernel(
    const __bf16* __restrict__ A1, const __bf16* __restrict__ W1, int K1,
    const __bf16* __restrict__ A2, const __bf16* __restrict__ W2, int K2,
    const float* __restrict__ bias, void* __restrict__ Cout, int N) {
  const int wave = threadIdx.x >> 5;
  const int lane = threadIdx.x & 31;
  const int wm = wave >> 2;   // 0..1
  const int wn = wave & 3;    // 0..3
  const int row0 = blockIdx.y * 64 + wm * 32;
  const int col0 = blockIdx.x * 256 + wn * 64;

  v8f c[2][4] = {};
  gemm_acc<2, 4>(A1, W1, K1, row0, col0, lane, c);
  if (K2 > 0) gemm_acc<2, 4>(A2, W2, K2, row0, col0, lane, c);

  const int l  = lane & 15;
  const int hi = lane >> 4;
#pragma unroll
  for (int i = 0; i < 2; ++i) {
#pragma unroll
    for (int j = 0; j < 4; ++j) {
      const int col   = col0 + j * 16 + l;
      const float bv  = bias[col];
      const int rbase = row0 + i * 16 + hi * 8;
#pragma unroll
      for (int r = 0; r < 8; ++r) {
        const float v = c[i][j][r] + bv;
        if (OUT_BF16)
          ((__bf16*)Cout)[(size_t)(rbase + r) * N + col] = (__bf16)v;
        else
          ((float*)Cout)[(size_t)(rbase + r) * N + col] = v;
      }
    }
  }
}

// ---------------------------------------------------------------------------
// Final layer: logits = c2 * Wfin^T + b_fin, then log_softmax over V=108.
// One block covers 64 rows x 128 cols (all of V), waves in 2x2-frag shape.
// Tile goes through LDS, then 64 threads do row-wise log-sum-exp.
// Grid: (1, B/64).
// ---------------------------------------------------------------------------
__global__ __launch_bounds__(256) void final_lsm_kernel(
    const __bf16* __restrict__ A, const __bf16* __restrict__ W,
    const float* __restrict__ bias, float* __restrict__ out, int K) {
  __shared__ float tile[64][132];  // padded stride to dodge bank conflicts
  const int wave = threadIdx.x >> 5;
  const int lane = threadIdx.x & 31;
  const int wm = wave >> 2;
  const int wn = wave & 3;
  const int row0 = blockIdx.y * 64 + wm * 32;
  const int col0 = wn * 32;

  v8f c[2][2] = {};
  gemm_acc<2, 2>(A, W, K, row0, col0, lane, c);

  const int l  = lane & 15;
  const int hi = lane >> 4;
#pragma unroll
  for (int i = 0; i < 2; ++i) {
#pragma unroll
    for (int j = 0; j < 2; ++j) {
      const int lc = col0 + j * 16 + l;
      const float bv = (lc < SZ_V) ? bias[lc] : 0.0f;
      const int lr = wm * 32 + i * 16 + hi * 8;
#pragma unroll
      for (int r = 0; r < 8; ++r) tile[lr + r][lc] = c[i][j][r] + bv;
    }
  }
  __syncthreads();

  if (threadIdx.x < 64) {
    const int row = threadIdx.x;
    float mx = -INFINITY;
    for (int v = 0; v < SZ_V; ++v) mx = fmaxf(mx, tile[row][v]);
    float s = 0.0f;
    for (int v = 0; v < SZ_V; ++v) s += __expf(tile[row][v] - mx);
    const float lz = mx + __logf(s);
    float* orow = out + (size_t)(blockIdx.y * 64 + row) * SZ_V;
    for (int v = 0; v < SZ_V; ++v) orow[v] = tile[row][v] - lz;
  }
}

// ---------------------------------------------------------------------------
// Pointwise LSTM cell: gates [B,4H] (i|f|g|o), c f32 in/out; emits bf16 c,h.
// ---------------------------------------------------------------------------
__device__ __forceinline__ float sigm(float x) { return 1.0f / (1.0f + __expf(-x)); }

__global__ __launch_bounds__(256) void lstm_cell_kernel(
    const float* __restrict__ gates, float* __restrict__ cf,
    __bf16* __restrict__ cbf, __bf16* __restrict__ hbf) {
  const int idx = blockIdx.x * 256 + threadIdx.x;  // over B*H
  const int b = idx >> 10;          // H = 1024
  const int h = idx & (SZ_H - 1);
  const float* g = gates + (size_t)b * SZ_G;
  const float gi = sigm(g[h]);
  const float gf = sigm(g[SZ_H + h]);
  const float gg = tanhf(g[2 * SZ_H + h]);
  const float go = sigm(g[3 * SZ_H + h]);
  const float cn = gf * cf[idx] + gi * gg;
  const float hn = go * tanhf(cn);
  cf[idx]  = cn;
  cbf[idx] = (__bf16)cn;
  hbf[idx] = (__bf16)hn;
}

// ---------------------------------------------------------------------------
// Setup kernels
// ---------------------------------------------------------------------------
// fp32 [N,K] -> bf16 [Npad,Kpad], zero-filled padding.
__global__ __launch_bounds__(256) void cvt_pad_kernel(
    const float* __restrict__ src, __bf16* __restrict__ dst,
    int N, int K, int Npad, int Kpad) {
  const size_t idx = (size_t)blockIdx.x * 256 + threadIdx.x;
  if (idx >= (size_t)Npad * Kpad) return;
  const int n = (int)(idx / Kpad);
  const int k = (int)(idx % Kpad);
  dst[idx] = (n < N && k < K) ? (__bf16)src[(size_t)n * K + k] : (__bf16)0.0f;
}

__global__ __launch_bounds__(256) void addvec_kernel(
    const float* __restrict__ a, const float* __restrict__ b,
    float* __restrict__ o, int n) {
  const int i = blockIdx.x * 256 + threadIdx.x;
  if (i < n) o[i] = a[i] + b[i];
}

__global__ __launch_bounds__(256) void zero_kernel(uint32_t* __restrict__ p, size_t nwords) {
  const size_t i = (size_t)blockIdx.x * 256 + threadIdx.x;
  if (i < nwords) p[i] = 0u;
}

// ---------------------------------------------------------------------------
// Host launcher
// ---------------------------------------------------------------------------
extern "C" void kernel_launch(void* const* d_in, const int* in_sizes, int n_in,
                              void* d_out, int out_size, void* d_ws, size_t ws_size,
                              hipStream_t stream) {
  (void)in_sizes; (void)n_in; (void)out_size; (void)ws_size;
  const float* x     = (const float*)d_in[0];
  const float* Wih1  = (const float*)d_in[1];
  const float* Whh1  = (const float*)d_in[2];
  const float* bih1  = (const float*)d_in[3];
  const float* bhh1  = (const float*)d_in[4];
  const float* Wmid  = (const float*)d_in[5];
  const float* bmid  = (const float*)d_in[6];
  const float* Wih2  = (const float*)d_in[7];
  const float* Whh2  = (const float*)d_in[8];
  const float* bih2  = (const float*)d_in[9];
  const float* bhh2  = (const float*)d_in[10];
  const float* Wfin  = (const float*)d_in[11];
  const float* bfin  = (const float*)d_in[12];
  float* out = (float*)d_out;
  char* ws = (char*)d_ws;

  // Workspace layout (all sizes are multiples of 256 bytes).
  size_t off = 0;
  auto take = [&](size_t bytes) {
    size_t o = off;
    off = (off + bytes + 255) & ~(size_t)255;
    return o;
  };
  const size_t o_wih1 = take((size_t)SZ_G * SZ_IP * 2);
  const size_t o_whh1 = take((size_t)SZ_G * SZ_H * 2);
  const size_t o_wmid = take((size_t)SZ_H * SZ_H * 2);
  const size_t o_wih2 = take((size_t)SZ_G * SZ_H * 2);
  const size_t o_whh2 = take((size_t)SZ_G * SZ_H * 2);
  const size_t o_wfin = take((size_t)SZ_VP * SZ_H * 2);
  const size_t o_xbf  = take((size_t)SZ_T * SZ_B * SZ_IP * 2);
  const size_t o_bs1  = take((size_t)SZ_G * 4);
  const size_t o_bs2  = take((size_t)SZ_G * 4);
  const size_t o_g1   = take((size_t)SZ_B * SZ_G * 4);
  const size_t o_g2   = take((size_t)SZ_B * SZ_G * 4);
  const size_t o_state = off;               // contiguous zero-init region
  const size_t o_c1f  = take((size_t)SZ_B * SZ_H * 4);
  const size_t o_c2f  = take((size_t)SZ_B * SZ_H * 4);
  const size_t o_h1bf = take((size_t)SZ_B * SZ_H * 2);
  const size_t o_c1bf = take((size_t)SZ_B * SZ_H * 2);
  const size_t o_h2bf = take((size_t)SZ_B * SZ_H * 2);
  const size_t o_c2bf = take((size_t)SZ_B * SZ_H * 2);
  const size_t state_bytes = off - o_state;
  const size_t o_mbf  = take((size_t)SZ_B * SZ_H * 2);

  __bf16* wih1bf = (__bf16*)(ws + o_wih1);
  __bf16* whh1bf = (__bf16*)(ws + o_whh1);
  __bf16* wmidbf = (__bf16*)(ws + o_wmid);
  __bf16* wih2bf = (__bf16*)(ws + o_wih2);
  __bf16* whh2bf = (__bf16*)(ws + o_whh2);
  __bf16* wfinbf = (__bf16*)(ws + o_wfin);
  __bf16* xbf    = (__bf16*)(ws + o_xbf);
  float*  bs1    = (float*)(ws + o_bs1);
  float*  bs2    = (float*)(ws + o_bs2);
  float*  g1     = (float*)(ws + o_g1);
  float*  g2     = (float*)(ws + o_g2);
  float*  c1f    = (float*)(ws + o_c1f);
  float*  c2f    = (float*)(ws + o_c2f);
  __bf16* h1bf   = (__bf16*)(ws + o_h1bf);
  __bf16* c1bf   = (__bf16*)(ws + o_c1bf);
  __bf16* h2bf   = (__bf16*)(ws + o_h2bf);
  __bf16* c2bf   = (__bf16*)(ws + o_c2bf);
  __bf16* mbf    = (__bf16*)(ws + o_mbf);

  auto blocks = [](size_t n) { return (unsigned)((n + 255) / 256); };

  // --- one-time setup (cheap; weights ~14M elems) ---
  cvt_pad_kernel<<<blocks((size_t)SZ_G * SZ_IP), 256, 0, stream>>>(Wih1, wih1bf, SZ_G, SZ_I, SZ_G, SZ_IP);
  cvt_pad_kernel<<<blocks((size_t)SZ_G * SZ_H), 256, 0, stream>>>(Whh1, whh1bf, SZ_G, SZ_H, SZ_G, SZ_H);
  cvt_pad_kernel<<<blocks((size_t)SZ_H * SZ_H), 256, 0, stream>>>(Wmid, wmidbf, SZ_H, SZ_H, SZ_H, SZ_H);
  cvt_pad_kernel<<<blocks((size_t)SZ_G * SZ_H), 256, 0, stream>>>(Wih2, wih2bf, SZ_G, SZ_H, SZ_G, SZ_H);
  cvt_pad_kernel<<<blocks((size_t)SZ_G * SZ_H), 256, 0, stream>>>(Whh2, whh2bf, SZ_G, SZ_H, SZ_G, SZ_H);
  cvt_pad_kernel<<<blocks((size_t)SZ_VP * SZ_H), 256, 0, stream>>>(Wfin, wfinbf, SZ_V, SZ_H, SZ_VP, SZ_H);
  cvt_pad_kernel<<<blocks((size_t)SZ_T * SZ_B * SZ_IP), 256, 0, stream>>>(x, xbf, SZ_T * SZ_B, SZ_I, SZ_T * SZ_B, SZ_IP);
  addvec_kernel<<<blocks(SZ_G), 256, 0, stream>>>(bih1, bhh1, bs1, SZ_G);
  addvec_kernel<<<blocks(SZ_G), 256, 0, stream>>>(bih2, bhh2, bs2, SZ_G);
  zero_kernel<<<blocks(state_bytes / 4), 256, 0, stream>>>((uint32_t*)(ws + o_state), state_bytes / 4);

  // --- recurrent loop ---
  const dim3 gridGates(SZ_G / 256, SZ_B / 64);  // 16 x 8 = 128 blocks
  const dim3 gridMid(SZ_H / 256, SZ_B / 64);    // 4 x 8
  const dim3 gridFin(1, SZ_B / 64);             // 8 blocks, each owns full rows
  const unsigned cellBlocks = blocks((size_t)SZ_B * SZ_H);

  for (int t = 0; t < SZ_T; ++t) {
    const __bf16* xt = xbf + (size_t)t * SZ_B * SZ_IP;
    // gates1 = x_t * Wih1^T + h1 * Whh1^T + (b_ih1 + b_hh1)
    gemm2_kernel<false><<<gridGates, 256, 0, stream>>>(
        xt, wih1bf, SZ_IP, h1bf, whh1bf, SZ_H, bs1, g1, SZ_G);
    lstm_cell_kernel<<<cellBlocks, 256, 0, stream>>>(g1, c1f, c1bf, h1bf);
    // m = c1 * Wmid^T + b_mid   (bf16 out, reference feeds the CELL state)
    gemm2_kernel<true><<<gridMid, 256, 0, stream>>>(
        c1bf, wmidbf, SZ_H, nullptr, nullptr, 0, bmid, mbf, SZ_H);
    // gates2 = m * Wih2^T + h2 * Whh2^T + (b_ih2 + b_hh2)
    gemm2_kernel<false><<<gridGates, 256, 0, stream>>>(
        mbf, wih2bf, SZ_H, h2bf, whh2bf, SZ_H, bs2, g2, SZ_G);
    lstm_cell_kernel<<<cellBlocks, 256, 0, stream>>>(g2, c2f, c2bf, h2bf);
    // out_t = log_softmax(c2 * Wfin^T + b_fin)
    final_lsm_kernel<<<gridFin, 256, 0, stream>>>(
        c2bf, wfinbf, bfin, out + (size_t)t * SZ_B * SZ_V, SZ_H);
  }
}
